// GTLayer_60318520705510
// MI455X (gfx1250) — matrix operand order repaired and checked by reading, not compiled, not run on hardware
//
#include <hip/hip_runtime.h>
#include <hip/hip_bf16.h>
#include <stdint.h>

#define NNODES  4096
#define NREL    5
#define NCH     2
#define NEDGES  131072
#define THRESH  0.05f

typedef __attribute__((ext_vector_type(16))) __bf16 v16bf;
typedef __attribute__((ext_vector_type(8)))  float  v8f;

__device__ __forceinline__ unsigned short f32_to_bf16_rne(float f) {
    unsigned int u = __float_as_uint(f);
    u += 0x7FFFu + ((u >> 16) & 1u);   // round to nearest even
    return (unsigned short)(u >> 16);
}

// ---------------------------------------------------------------- softmax ----
__global__ void softmax_rows_kernel(const float* __restrict__ w1,
                                    const float* __restrict__ w2,
                                    float* __restrict__ Fws,
                                    float* __restrict__ outTail) {
    if (threadIdx.x == 0) {
        const float* w[2] = {w1, w2};
        for (int s = 0; s < 2; ++s) {
            for (int c = 0; c < NCH; ++c) {
                float mx = -1e30f;
                for (int r = 0; r < NREL; ++r) mx = fmaxf(mx, w[s][c*NREL + r]);
                float e[NREL], sum = 0.f;
                for (int r = 0; r < NREL; ++r) { e[r] = __expf(w[s][c*NREL + r] - mx); sum += e[r]; }
                float inv = 1.0f / sum;
                for (int r = 0; r < NREL; ++r) {
                    float f = e[r] * inv;
                    Fws[s*10 + c*NREL + r]     = f;
                    outTail[s*10 + c*NREL + r] = f;
                }
            }
        }
    }
}

// ------------------------------------------------------------------- zero ----
__global__ void zero_f32_kernel(float4* __restrict__ p, int n4) {
    int i = blockIdx.x * blockDim.x + threadIdx.x;
    int stride = gridDim.x * blockDim.x;
    float4 z = {0.f, 0.f, 0.f, 0.f};
    for (; i < n4; i += stride) p[i] = z;
}

// ---------------------------------------------------------------- scatter ----
// Fused dense_adj + channel mixing: acc[c][src][dst] += F[c,r] * ev[r,e].
__global__ void scatter_kernel(const int*   __restrict__ ei,
                               const float* __restrict__ ev,
                               const float* __restrict__ F,   // [2][NREL]
                               float*       __restrict__ acc) {
    int idx = blockIdx.x * blockDim.x + threadIdx.x;
    if (idx >= NREL * NEDGES) return;
    int r = idx >> 17;             // / NEDGES (= 2^17)
    int e = idx & (NEDGES - 1);
    int src = ei[(size_t)r * 2 * NEDGES + e];
    int dst = ei[(size_t)r * 2 * NEDGES + NEDGES + e];
    float v = ev[(size_t)r * NEDGES + e];
    size_t cell = (size_t)src * NNODES + dst;
    atomicAdd(acc + cell, F[r] * v);                                    // channel 0
    atomicAdd(acc + (size_t)NNODES * NNODES + cell, F[NREL + r] * v);   // channel 1
}

// ---------------------------------------------------------------- convert ----
__global__ void cvt_bf16_kernel(const float4* __restrict__ in,
                                ushort4* __restrict__ out, int n4) {
    int i = blockIdx.x * blockDim.x + threadIdx.x;
    int stride = gridDim.x * blockDim.x;
    for (; i < n4; i += stride) {
        float4 f = in[i];
        ushort4 o;
        o.x = f32_to_bf16_rne(f.x); o.y = f32_to_bf16_rne(f.y);
        o.z = f32_to_bf16_rne(f.z); o.w = f32_to_bf16_rne(f.w);
        out[i] = o;
    }
}

// ------------------------------------------------------------------- GEMM ----
// Per channel: H = A1 (MxK) * A2 (KxN), bf16 in / f32 accumulate, fused
// threshold. Block = 256 threads (8 waves) -> 128x64 output tile.
// Double-buffered LDS software pipeline: global loads for panel k+1 issued
// before the WMMAs of panel k; one barrier per K-step.
#define TM 128
#define TN 64
#define TK 32
#define LDSROW 80                  // 32 bf16 = 64B + 16B pad (16B-aligned)
#define SM_A (TM * LDSROW)         // 10240 B
#define SM_B (TN * LDSROW)         //  5120 B
#define BUFB (SM_A + SM_B)         // 15360 B per pipeline stage

__global__ void __launch_bounds__(256)
gemm_bf16_kernel(const unsigned short* __restrict__ Abf,   // [C][M][K]
                 const unsigned short* __restrict__ Bbf,   // [C][K][N]
                 float* __restrict__ H) {
    __shared__ __align__(16) unsigned char smem[2 * BUFB];

    const int n0   = blockIdx.x * TN;
    const int m0   = blockIdx.y * TM;
    const int c    = blockIdx.z;
    const int tid  = threadIdx.x;
    const int wave = tid >> 5;
    const int lane = tid & 31;
    const int l16  = lane & 15;
    const int hi   = lane >> 4;        // 0: lanes 0-15, 1: lanes 16-31

    const size_t NN = (size_t)NNODES * NNODES;
    const uint4* Ag = (const uint4*)(Abf + (size_t)c * NN);
    const uint4* Bg = (const uint4*)(Bbf + (size_t)c * NN);

    // ---- constant per-thread staging geometry ----
    // A tile [128 x 32] bf16 = 512 x 16B chunks; thread covers chunks tid and tid+256.
    const int aRow0 = tid >> 2;                    // rows 0..63
    const int aRow1 = (tid + 256) >> 2;            // rows 64..127
    const int aC4   = tid & 3;
    // B tile [32 x 64] bf16, read row-major coalesced, stored transposed Bt[n][k].
    const int kr = tid >> 3;                       // 0..31
    const int nb = (tid & 7) * 8;                  // 0..56
    // global uint4 indices (8 bf16 per uint4); advance per K-step.
    size_t giA0 = (size_t)(m0 + aRow0) * (NNODES >> 3) + aC4;
    size_t giA1 = (size_t)(m0 + aRow1) * (NNODES >> 3) + aC4;
    size_t giB  = (size_t)kr * (NNODES >> 3) + (n0 >> 3) + (tid & 7);
    const size_t dA = TK >> 3;                     // 4
    const size_t dB = (size_t)TK * (NNODES >> 3);  // 16384
    // constant LDS store offsets (within a stage)
    const int sA0 = aRow0 * LDSROW + aC4 * 16;
    const int sA1 = aRow1 * LDSROW + aC4 * 16;

    v8f zero = {};
    v8f acc[4] = {zero, zero, zero, zero};

    // ---- prologue: fetch panel 0 into registers ----
    uint4 ra0 = Ag[giA0], ra1 = Ag[giA1], rb = Bg[giB];
    giA0 += dA; giA1 += dA; giB += dB;

    int p = 0;
    for (int kk = 0; kk < NNODES; kk += TK) {
        unsigned char* smA = smem + p * BUFB;
        unsigned char* smB = smA + SM_A;

        // ---- store current panel registers into stage p ----
        *(uint4*)(smA + sA0) = ra0;
        *(uint4*)(smA + sA1) = ra1;
        {
            const unsigned short* sv = (const unsigned short*)&rb;
#pragma unroll
            for (int j = 0; j < 8; ++j)
                *(unsigned short*)(smB + (nb + j) * LDSROW + kr * 2) = sv[j];
        }
        __syncthreads();

        // ---- issue next panel's global loads (overlap with WMMAs below) ----
        if (kk + TK < NNODES) {
            ra0 = Ag[giA0]; ra1 = Ag[giA1]; rb = Bg[giB];
            giA0 += dA; giA1 += dA; giB += dB;
        }

        // ---- fragments: A row strip + all four B column blocks, then a
        //      back-to-back WMMA burst (single LDS wait) ----
        union { uint4 u[2]; v16bf v; } af, b0, b1, b2, b3;
        const unsigned char* arow = smA + (wave * 16 + l16) * LDSROW;
        af.u[0] = *(const uint4*)(arow + hi * 16);
        af.u[1] = *(const uint4*)(arow + 32 + hi * 16);
        const unsigned char* brow = smB + l16 * LDSROW + hi * 16;
        b0.u[0] = *(const uint4*)(brow);
        b0.u[1] = *(const uint4*)(brow + 32);
        b1.u[0] = *(const uint4*)(brow + 16 * LDSROW);
        b1.u[1] = *(const uint4*)(brow + 16 * LDSROW + 32);
        b2.u[0] = *(const uint4*)(brow + 32 * LDSROW);
        b2.u[1] = *(const uint4*)(brow + 32 * LDSROW + 32);
        b3.u[0] = *(const uint4*)(brow + 48 * LDSROW);
        b3.u[1] = *(const uint4*)(brow + 48 * LDSROW + 32);

        acc[0] = __builtin_amdgcn_wmma_f32_16x16x32_bf16(false, af.v, false, b0.v, (short)0, acc[0], false, false);
        acc[1] = __builtin_amdgcn_wmma_f32_16x16x32_bf16(false, af.v, false, b1.v, (short)0, acc[1], false, false);
        acc[2] = __builtin_amdgcn_wmma_f32_16x16x32_bf16(false, af.v, false, b2.v, (short)0, acc[2], false, false);
        acc[3] = __builtin_amdgcn_wmma_f32_16x16x32_bf16(false, af.v, false, b3.v, (short)0, acc[3], false, false);

        p ^= 1;
    }

    // ---- writeout with fused threshold; D layout: VGPR i, lanes 0-15 ->
    //      (M=i, N=lane), lanes 16-31 -> (M=i+8, N=lane-16) ----
    float* Hc = H + (size_t)c * NN;
    const int rowb = m0 + wave * 16 + hi * 8;
#pragma unroll
    for (int ni = 0; ni < 4; ++ni) {
        int col = n0 + ni * 16 + l16;
#pragma unroll
        for (int i = 0; i < 8; ++i) {
            float v = acc[ni][i];
            Hc[(size_t)(rowb + i) * NNODES + col] = (v > THRESH) ? v : 0.0f;
        }
    }
}

// ----------------------------------------------------------------- launch ----
extern "C" void kernel_launch(void* const* d_in, const int* in_sizes, int n_in,
                              void* d_out, int out_size, void* d_ws, size_t ws_size,
                              hipStream_t stream) {
    const int*   ei = (const int*)d_in[0];
    const float* ev = (const float*)d_in[1];
    const float* w1 = (const float*)d_in[2];
    const float* w2 = (const float*)d_in[3];
    float* out = (float*)d_out;

    const size_t NN = (size_t)NNODES * NNODES;
    // workspace layout: [0,128MB) f32 accumulator (reused for A1 then A2),
    // [128MB,192MB) A1 bf16, [192MB,256MB) A2 bf16, then 20 softmax floats.
    float*          accf = (float*)d_ws;
    unsigned short* A1b  = (unsigned short*)((char*)d_ws + 2 * NN * 4);
    unsigned short* A2b  = (unsigned short*)((char*)d_ws + 2 * NN * 4 + 2 * NN * 2);
    float*          Fws  = (float*)((char*)d_ws + 2 * NN * 4 + 4 * NN * 2);

    const int n4 = (int)(2 * NN / 4);          // float4 count = 8,388,608
    const int scatterBlocks = (NREL * NEDGES + 255) / 256;

    softmax_rows_kernel<<<1, 32, 0, stream>>>(w1, w2, Fws, out + 2 * NN);

    // channel-mixed A1 = softmax(w1) . A  (scatter in f32, round to bf16)
    zero_f32_kernel<<<8192, 256, 0, stream>>>((float4*)accf, n4);
    scatter_kernel<<<scatterBlocks, 256, 0, stream>>>(ei, ev, Fws, accf);
    cvt_bf16_kernel<<<8192, 256, 0, stream>>>((const float4*)accf, (ushort4*)A1b, n4);

    // channel-mixed A2 = softmax(w2) . A
    zero_f32_kernel<<<8192, 256, 0, stream>>>((float4*)accf, n4);
    scatter_kernel<<<scatterBlocks, 256, 0, stream>>>(ei, ev, Fws + 10, accf);
    cvt_bf16_kernel<<<8192, 256, 0, stream>>>((const float4*)accf, (ushort4*)A2b, n4);

    // per-channel GEMM + threshold -> H
    dim3 grid(NNODES / TN, NNODES / TM, NCH);
    gemm_bf16_kernel<<<grid, 256, 0, stream>>>(A1b, A2b, out);
}
